// UniversalMoERouter_39539468927389
// MI455X (gfx1250) — matrix operand order repaired and checked by reading, not compiled, not run on hardware
//
#include <hip/hip_runtime.h>
#include <hip/hip_bf16.h>

// ---------------- problem constants ----------------
constexpr int Bc  = 4;
constexpr int Sc  = 512;
constexpr int Hc  = 2048;
constexpr int Ec  = 16;
constexpr int Dc  = 128;
constexpr int GHc = 2048;      // E*D
constexpr int G3c = 6144;      // 3*GH
constexpr int NTOK = Bc * Sc;  // 2048

// ---------------- types ----------------
typedef __attribute__((ext_vector_type(16))) _Float16 v16h;
typedef __attribute__((ext_vector_type(8)))  float    v8f;
typedef __attribute__((ext_vector_type(8)))  _Float16 v8h;
typedef __attribute__((ext_vector_type(4)))  _Float16 v4h;

struct __attribute__((aligned(16))) F4 { float x, y, z, w; };
union AF { v16h h; F4 f[2]; };

// ---------------- async global->LDS copy (gfx1250 ASYNCcnt path) ----------------
#if defined(__has_builtin)
#if __has_builtin(__builtin_amdgcn_global_load_async_to_lds_b128) && \
    __has_builtin(__builtin_amdgcn_s_wait_asynccnt)
#define USE_ASYNC_LDS 1
#endif
#endif
#ifndef USE_ASYNC_LDS
#define USE_ASYNC_LDS 0
#endif

#if USE_ASYNC_LDS
typedef int v4i __attribute__((vector_size(16)));
typedef __attribute__((address_space(1))) v4i gv4i;   // global v4i
typedef __attribute__((address_space(3))) v4i lv4i;   // LDS v4i
#endif

__device__ __forceinline__ void lds_copy16(void* l, const void* g) {
#if USE_ASYNC_LDS
  __builtin_amdgcn_global_load_async_to_lds_b128(
      (gv4i*)(g), (lv4i*)(l), 0, 0);
#else
  *(F4*)l = *(const F4*)g;
#endif
}
__device__ __forceinline__ void lds_copy_wait() {
#if USE_ASYNC_LDS
  __builtin_amdgcn_s_wait_asynccnt(0);
#endif
}

// ---------------- init scratch state ----------------
__global__ void init_kernel(float* penAcc, float* lossAcc,
                            unsigned* barCount, unsigned* barGen) {
  if (threadIdx.x == 0) {
    *penAcc = 0.f; *lossAcc = 0.f; *barCount = 0u; *barGen = 0u;
  }
}

// ---------------- f32 -> f16 convert (vectorized) ----------------
__global__ __launch_bounds__(256)
void cvt_f16_kernel(const float* __restrict__ src, _Float16* __restrict__ dst, int n4) {
  int i = blockIdx.x * 256 + threadIdx.x;
  if (i >= n4) return;
  F4 v = ((const F4*)src)[i];
  v4h h; h[0] = (_Float16)v.x; h[1] = (_Float16)v.y;
  h[2] = (_Float16)v.z; h[3] = (_Float16)v.w;
  ((v4h*)dst)[i] = h;
}

// ---------------- tiled transpose + convert: src[rows][cols] -> dst[cols][rows] ----
__global__ __launch_bounds__(256)
void transpose_cvt_kernel(const float* __restrict__ src, _Float16* __restrict__ dst,
                          int rows, int cols) {
  __shared__ float tile[32][33];
  const int bx = blockIdx.x * 32;           // col base
  const int by = blockIdx.y * 32;           // row base
  const int tx = threadIdx.x & 31, ty = threadIdx.x >> 5;  // 32 x 8
#pragma unroll
  for (int i = 0; i < 32; i += 8)
    tile[ty + i][tx] = src[(size_t)(by + ty + i) * cols + bx + tx];
  __syncthreads();
#pragma unroll
  for (int i = 0; i < 32; i += 8)
    dst[(size_t)(bx + ty + i) * rows + by + tx] = (_Float16)tile[tx][ty + i];
}

// ---------------- pack w_hh into interleaved [k][j][(r,z,n,0)] f16 ----------------
__global__ __launch_bounds__(256)
void pack_whh_kernel(const float* __restrict__ w_hh, _Float16* __restrict__ dst) {
  int idx = blockIdx.x * 256 + threadIdx.x;      // idx = k*GH + j
  if (idx >= GHc * GHc) return;
  int k = idx / GHc, j = idx - k * GHc;
  v4h v;
  v[0] = (_Float16)w_hh[((size_t)(0 * GHc + j)) * GHc + k];
  v[1] = (_Float16)w_hh[((size_t)(1 * GHc + j)) * GHc + k];
  v[2] = (_Float16)w_hh[((size_t)(2 * GHc + j)) * GHc + k];
  v[3] = (_Float16)0.f;
  ((v4h*)dst)[idx] = v;
}

// ---------------- WMMA GEMM: C[M,N] = A[M,K](f16,row) * B[K,N](f16,row) --------
// Block tile 64(M) x 128(N), BK=32. 8 waves: wave = (wm 0..3, wg 0..1),
// each wave computes 16x64 of C via 4 chained WMMAs sharing one A fragment
// (reuse_a hint set on the 2nd..4th identical back-to-back WMMAs).
// mode 0: store C (f32).  mode 1: accumulate sum((C - I)^2) into lossAcc.
__global__ __launch_bounds__(256)
void gemm_f16_kernel(const _Float16* __restrict__ A, const _Float16* __restrict__ Bk,
                     float* __restrict__ C, float* __restrict__ lossAcc,
                     int M, int N, int K, int mode) {
  __shared__ __align__(16) _Float16 As[64 * 40];
  __shared__ __align__(16) _Float16 Bs[128 * 40];

  const int tid  = threadIdx.x;
  const int lane = tid & 31, wave = tid >> 5;
  const int wm = wave & 3, wg = wave >> 2;            // 4 x 2 wave grid
  const int m0 = blockIdx.y * 64, n0 = blockIdx.x * 128;

  v8f acc[4] = {{}, {}, {}, {}};
  const int ar = tid >> 2, ac = (tid & 3) * 8;        // A staging: 64 rows x 32 cols
  const int br = tid >> 3, bc = (tid & 7) * 16;       // B staging: 32 rows x 128 cols

  for (int kk = 0; kk < K; kk += 32) {
    // stage A tile [64 x 32] straight into LDS (async when available)
    lds_copy16(As + ar * 40 + ac, A + (size_t)(m0 + ar) * K + kk + ac);
    // stage B tile transposed: Bs[n][k] = B[kk+k][n0+n]
    const _Float16* bsrc = Bk + (size_t)(kk + br) * N + n0 + bc;
    v8h bv0 = *(const v8h*)bsrc;
    v8h bv1 = *(const v8h*)(bsrc + 8);
#pragma unroll
    for (int u = 0; u < 8; ++u) {
      Bs[(bc + u) * 40 + br]     = bv0[u];
      Bs[(bc + 8 + u) * 40 + br] = bv1[u];
    }
    if (kk + 32 < K) {
      __builtin_prefetch(A + (size_t)(m0 + ar) * K + kk + 32 + ac, 0, 1);
      __builtin_prefetch(Bk + (size_t)(kk + 32 + br) * N + n0 + bc, 0, 1);
    }
    lds_copy_wait();
    __syncthreads();

    // A fragment: lane m = lane%16, k-half per lane>>4 (ISA 16-bit A 16x32 layout)
    AF a;
    const int mrow = wm * 16 + (lane & 15);
    const int k0 = (lane >> 4) * 8;
    a.f[0] = *(const F4*)(As + mrow * 40 + k0);
    a.f[1] = *(const F4*)(As + mrow * 40 + 16 + k0);
    // B fragments: lane n = lane%16, same k-half layout as A
    const int kg = (lane >> 4) * 8;
    AF b[4];
#pragma unroll
    for (int j = 0; j < 4; ++j) {
      const int nr = wg * 64 + j * 16 + (lane & 15);
      b[j].f[0] = *(const F4*)(Bs + nr * 40 + kg);
      b[j].f[1] = *(const F4*)(Bs + nr * 40 + kg + 16);
    }
    // 4 back-to-back WMMAs sharing A: set reuse_a on the trailing identical ops
    acc[0] = __builtin_amdgcn_wmma_f32_16x16x32_f16(false, a.h, false, b[0].h,
                                                    (short)0, acc[0], false, false);
    acc[1] = __builtin_amdgcn_wmma_f32_16x16x32_f16(false, a.h, false, b[1].h,
                                                    (short)0, acc[1], true, false);
    acc[2] = __builtin_amdgcn_wmma_f32_16x16x32_f16(false, a.h, false, b[2].h,
                                                    (short)0, acc[2], true, false);
    acc[3] = __builtin_amdgcn_wmma_f32_16x16x32_f16(false, a.h, false, b[3].h,
                                                    (short)0, acc[3], true, false);
    __syncthreads();
  }

  // C layout: VGPR i -> row i + 8*(lane>>4), col lane&15
  const int ml = (lane >> 4) * 8;
  const int nl = lane & 15;
  if (mode == 0) {
#pragma unroll
    for (int i = 0; i < 8; ++i) {
      const int m = m0 + wm * 16 + ml + i;
#pragma unroll
      for (int j = 0; j < 4; ++j)
        C[(size_t)m * N + n0 + wg * 64 + j * 16 + nl] = acc[j][i];
    }
  } else {
    float ls = 0.f;
#pragma unroll
    for (int i = 0; i < 8; ++i) {
      const int m = m0 + wm * 16 + ml + i;
#pragma unroll
      for (int j = 0; j < 4; ++j) {
        const int n = n0 + wg * 64 + j * 16 + nl;
        float v = acc[j][i] - ((m == n) ? 1.f : 0.f);
        ls += v * v;
      }
    }
    atomicAdd(lossAcc, ls);
  }
}

// ---------------- device-wide barrier (persistent kernel) ----------------
__device__ inline void grid_barrier(volatile unsigned* count, volatile unsigned* gen,
                                    unsigned nb) {
  __syncthreads();
  if (threadIdx.x == 0) {
    __threadfence();
    unsigned g = *gen;
    if (atomicAdd((unsigned*)count, 1u) == nb - 1u) {
      *count = 0u;
      __threadfence();
      atomicAdd((unsigned*)gen, 1u);
    } else {
      while (*gen == g) { __builtin_amdgcn_s_sleep(1); }
    }
    __threadfence();
  }
  __syncthreads();
}

// ---------------- GRU recurrent scan (persistent, 8 blocks) ----------------
// xi: [tok=b*S+s][3GH] f32 precomputed input projection.
// whhP: [k][j][(r,z,n,pad)] f16 -> one coalesced 8B load per k per thread.
__global__ __launch_bounds__(256)
void gru_scan_kernel(const float* __restrict__ xi, const _Float16* __restrict__ whhP,
                     const float* __restrict__ hn_in, float* __restrict__ rout,
                     float* __restrict__ h0, float* __restrict__ h1,
                     float* __restrict__ hn_out,
                     unsigned* barCount, unsigned* barGen, int nBlocks) {
  __shared__ __align__(16) float hsh[Bc * GHc];   // 32 KB of the 320 KB WGP LDS
  const int tid = threadIdx.x;
  const int j = blockIdx.x * 256 + tid;           // output channel 0..GH-1
  const int nthr = nBlocks * 256;

  for (int i = j; i < Bc * GHc; i += nthr) h0[i] = hn_in[i];
  grid_barrier(barCount, barGen, (unsigned)nBlocks);

  const v4h* wp = (const v4h*)whhP + j;           // stride GHc per k

  for (int s = 0; s < Sc; ++s) {
    const float* hc = (s & 1) ? h1 : h0;
    float*       hx = (s & 1) ? h0 : h1;
    // stage h broadcast copy into LDS (async 16B chunks when available)
    for (int i = tid * 4; i < Bc * GHc; i += 1024)
      lds_copy16(hsh + i, hc + i);
    lds_copy_wait();
    __syncthreads();

    float dr[Bc] = {}, dz[Bc] = {}, dn[Bc] = {};
    for (int k = 0; k < GHc; ++k) {
      v4h wv = wp[(size_t)k * GHc];
      float wr = (float)wv[0], wz = (float)wv[1], wn = (float)wv[2];
#pragma unroll
      for (int b = 0; b < Bc; ++b) {
        float hv = hsh[b * GHc + k];
        dr[b] = fmaf(hv, wr, dr[b]);
        dz[b] = fmaf(hv, wz, dz[b]);
        dn[b] = fmaf(hv, wn, dn[b]);
      }
    }
#pragma unroll
    for (int b = 0; b < Bc; ++b) {
      const size_t t = (size_t)b * Sc + s;
      const float* xg = xi + t * G3c;
      float r = 1.f / (1.f + __expf(-(xg[j] + dr[b])));
      float z = 1.f / (1.f + __expf(-(xg[GHc + j] + dz[b])));
      float n = tanhf(xg[2 * GHc + j] + r * dn[b]);
      float hold = hsh[b * GHc + j];
      float hnew = (1.f - z) * n + z * hold;
      hx[b * GHc + j] = hnew;
      rout[t * GHc + j] = hnew;
    }
    grid_barrier(barCount, barGen, (unsigned)nBlocks);
  }
  // after 512 steps the final h is in h0 (512 is even)
  for (int i = j; i < Bc * GHc; i += nthr) hn_out[i] = h0[i];
}

// ---------------- per-token: normalize, WMMA Gram penalty, cosine sims ----------
__global__ __launch_bounds__(256)
void token_post_kernel(const float* __restrict__ rout, const float* __restrict__ expr,
                       float* __restrict__ cosOut, float* __restrict__ penAcc) {
  __shared__ __align__(16) _Float16 R[8][Ec * Dc];  // 8 waves * 4KB = 32 KB
  const int wv = threadIdx.x >> 5, lane = threadIdx.x & 31;
  const int t = blockIdx.x * 8 + wv;
  const float* rp = rout + (size_t)t * GHc;

  // per-expert inverse L2 norms (row of 128, 4 elems/lane + xor reduce)
  float rinv[Ec];
#pragma unroll
  for (int e = 0; e < Ec; ++e) {
    F4 v = *(const F4*)(rp + e * Dc + lane * 4);
    float s2 = v.x * v.x + v.y * v.y + v.z * v.z + v.w * v.w;
#pragma unroll
    for (int m = 16; m >= 1; m >>= 1) s2 += __shfl_xor(s2, m);
    rinv[e] = 1.f / fmaxf(sqrtf(s2), 1e-12f);
  }
  // stage normalized rows as f16 for WMMA
#pragma unroll
  for (int e = 0; e < Ec; ++e) {
    F4 v = *(const F4*)(rp + e * Dc + lane * 4);
    float ri = rinv[e];
    v4h hv; hv[0] = (_Float16)(v.x * ri); hv[1] = (_Float16)(v.y * ri);
    hv[2] = (_Float16)(v.z * ri); hv[3] = (_Float16)(v.w * ri);
    *(v4h*)&R[wv][e * Dc + lane * 4] = hv;
  }
  __syncthreads();

  // gram = Rn * Rn^T via 4 chained 16x16x32 WMMAs
  v8f acc = {};
#pragma unroll
  for (int kk = 0; kk < Dc; kk += 32) {
    AF a, b;
    const int mr = lane & 15;
    const int k0 = kk + (lane >> 4) * 8;
    a.f[0] = *(const F4*)&R[wv][mr * Dc + k0];
    a.f[1] = *(const F4*)&R[wv][mr * Dc + k0 + 16];
    const int kg = kk + (lane >> 4) * 8;
    b.f[0] = *(const F4*)&R[wv][mr * Dc + kg];
    b.f[1] = *(const F4*)&R[wv][mr * Dc + kg + 16];
    acc = __builtin_amdgcn_wmma_f32_16x16x32_f16(false, a.h, false, b.h,
                                                 (short)0, acc, false, false);
  }

  // penalty: sum_e || normalize(row_e(gram - I)) ||^2
  float tokPen = 0.f;
  const int f = lane & 15;
#pragma unroll
  for (int i = 0; i < 8; ++i) {
    const int e = i + 8 * (lane >> 4);
    float dv = acc[i] - ((e == f) ? 1.f : 0.f);
    float s2 = dv * dv;
    s2 += __shfl_xor(s2, 1); s2 += __shfl_xor(s2, 2);
    s2 += __shfl_xor(s2, 4); s2 += __shfl_xor(s2, 8);
    float den = fmaxf(sqrtf(s2), 1e-12f);
    tokPen += s2 / (den * den);
  }
  tokPen += __shfl_xor(tokPen, 16);
  if (lane == 0) atomicAdd(penAcc, tokPen);

  // cosine similarities: 1 - <expr_e, rn_e> / (max(|expr_e|,1e-8)*max(|rn_e|,1e-8))
#pragma unroll
  for (int e = 0; e < Ec; ++e) {
    F4 xv = *(const F4*)(expr + (size_t)t * GHc + e * Dc + lane * 4);
    F4 rv = *(const F4*)(rp + e * Dc + lane * 4);
    float ri = rinv[e];
    float r0 = rv.x * ri, r1 = rv.y * ri, r2 = rv.z * ri, r3 = rv.w * ri;
    float dot = xv.x * r0 + xv.y * r1 + xv.z * r2 + xv.w * r3;
    float en2 = xv.x * xv.x + xv.y * xv.y + xv.z * xv.z + xv.w * xv.w;
    float rn2 = r0 * r0 + r1 * r1 + r2 * r2 + r3 * r3;
#pragma unroll
    for (int m = 16; m >= 1; m >>= 1) {
      dot += __shfl_xor(dot, m);
      en2 += __shfl_xor(en2, m);
      rn2 += __shfl_xor(rn2, m);
    }
    if (lane == 0) {
      float cs = dot / (fmaxf(sqrtf(en2), 1e-8f) * fmaxf(sqrtf(rn2), 1e-8f));
      cosOut[(size_t)t * Ec + e] = 1.f - cs;
    }
  }
}

// ---------------- finalize scalar outputs ----------------
__global__ void finalize_kernel(const float* penAcc, const float* lossAcc,
                                float* outPen, float* outLoss) {
  if (threadIdx.x == 0) {
    *outPen  = *penAcc  / (float)NTOK;
    *outLoss = *lossAcc / ((float)GHc * (float)GHc);
  }
}

// ---------------- top-2 + softmax + EMA adjustment ----------------
__global__ __launch_bounds__(256)
void topk_kernel(const float* __restrict__ cosSim, const float* __restrict__ penScalar,
                 const float* __restrict__ ema,
                 float* __restrict__ outMult, float* __restrict__ outSel) {
  const int t = blockIdx.x * 256 + threadIdx.x;
  if (t >= NTOK) return;
  const float scale = 1.f + *penScalar;
  float v1 = -3.4e38f, v2 = -3.4e38f; int i1 = 0, i2 = 0;
#pragma unroll
  for (int e = 0; e < Ec; ++e) {
    float v = cosSim[(size_t)t * Ec + e] * scale;
    if (v > v1)      { v2 = v1; i2 = i1; v1 = v; i1 = e; }
    else if (v > v2) { v2 = v;  i2 = e; }
  }
  float e2 = __expf(v2 - v1);
  float inv = 1.f / (1.f + e2);
  float p1 = inv, p2 = e2 * inv;
  float tot = 0.f;
#pragma unroll
  for (int e = 0; e < Ec; ++e) tot += ema[e];
  float adj1 = 0.f, adj2 = 0.f;
  if (tot > 0.f) {
    float it = 1.f / fmaxf(tot, 1e-12f);
    adj1 = ema[i1] * it * 0.01f * (float)Ec;
    adj2 = ema[i2] * it * 0.01f * (float)Ec;
  }
  outMult[t * 2 + 0] = p1 - adj1;
  outMult[t * 2 + 1] = p2 - adj2;
  outSel[t * 2 + 0] = (float)i1;
  outSel[t * 2 + 1] = (float)i2;
}

// ---------------- host launch ----------------
extern "C" void kernel_launch(void* const* d_in, const int* in_sizes, int n_in,
                              void* d_out, int out_size, void* d_ws, size_t ws_size,
                              hipStream_t stream) {
  (void)in_sizes; (void)n_in; (void)out_size; (void)ws_size;
  const float* x      = (const float*)d_in[0];
  const float* hn     = (const float*)d_in[1];
  const float* w_ih   = (const float*)d_in[2];
  const float* w_hh   = (const float*)d_in[3];
  const float* w_expr = (const float*)d_in[4];
  const float* ema    = (const float*)d_in[5];
  float* out = (float*)d_out;

  // output regions (flat concat in return order, all f32)
  float* outMult = out;                                   // [2048,2]
  float* outSel  = out + 4096;                            // [2048,2]
  float* outExpr = out + 8192;                            // [B,S,E,D]
  float* outHn   = outExpr + (size_t)NTOK * GHc;          // [1,B,GH]
  float* outPen  = outHn + (size_t)Bc * GHc;              // scalar
  float* outCos  = outPen + 1;                            // [B,S,E]
  float* outLoss = outCos + (size_t)NTOK * Ec;            // scalar

  // workspace carve-out
  char* ws = (char*)d_ws;
  size_t off = 0;
  auto carve = [&](size_t bytes) -> char* {
    char* p = ws + off;
    off = (off + bytes + 255) & ~(size_t)255;
    return p;
  };
  float*    xi      = (float*)   carve((size_t)NTOK * G3c * 4);   // input projection
  float*    routbuf = (float*)   carve((size_t)NTOK * GHc * 4);   // routing logits
  _Float16* xbf     = (_Float16*)carve((size_t)NTOK * Hc * 2);
  _Float16* wihT    = (_Float16*)carve((size_t)Hc * G3c * 2);
  _Float16* whhP    = (_Float16*)carve((size_t)GHc * GHc * 4 * 2);
  _Float16* wex     = (_Float16*)carve((size_t)Hc * GHc * 2);
  _Float16* wexT    = (_Float16*)carve((size_t)GHc * Hc * 2);
  float*    h0buf   = (float*)   carve((size_t)Bc * GHc * 4);
  float*    h1buf   = (float*)   carve((size_t)Bc * GHc * 4);
  float*    penAcc  = (float*)   carve(256);
  float*    lossAcc = (float*)   carve(256);
  unsigned* barCnt  = (unsigned*)carve(256);
  unsigned* barGen  = (unsigned*)carve(256);

  init_kernel<<<1, 64, 0, stream>>>(penAcc, lossAcc, barCnt, barGen);

  // f16 conversions / transposes / packs
  cvt_f16_kernel<<<(NTOK * Hc / 4 + 255) / 256, 256, 0, stream>>>(x, xbf, NTOK * Hc / 4);
  cvt_f16_kernel<<<(Hc * GHc / 4 + 255) / 256, 256, 0, stream>>>(w_expr, wex, Hc * GHc / 4);
  transpose_cvt_kernel<<<dim3(Hc / 32, G3c / 32), 256, 0, stream>>>(w_ih, wihT, G3c, Hc);
  transpose_cvt_kernel<<<dim3(GHc / 32, Hc / 32), 256, 0, stream>>>(w_expr, wexT, Hc, GHc);
  pack_whh_kernel<<<(GHc * GHc + 255) / 256, 256, 0, stream>>>(w_hh, whhP);

  // xi = x @ w_ih^T   [2048 x 6144]
  gemm_f16_kernel<<<dim3(G3c / 128, NTOK / 64), 256, 0, stream>>>(
      xbf, wihT, xi, nullptr, NTOK, G3c, Hc, 0);
  // expression_logits = x @ w_expr   [2048 x 2048] -> straight to d_out
  gemm_f16_kernel<<<dim3(GHc / 128, NTOK / 64), 256, 0, stream>>>(
      xbf, wex, outExpr, nullptr, NTOK, GHc, Hc, 0);
  // expression_loss = mean((w_expr^T w_expr - I)^2), fused reduction epilogue
  gemm_f16_kernel<<<dim3(GHc / 128, GHc / 64), 256, 0, stream>>>(
      wexT, wex, nullptr, lossAcc, GHc, GHc, Hc, 1);

  // sequential GRU scan (persistent, 8 co-resident blocks)
  gru_scan_kernel<<<8, 256, 0, stream>>>(xi, whhP, hn, routbuf, h0buf, h1buf,
                                         outHn, barCnt, barGen, 8);

  // per-token Gram penalty (WMMA) + cosine similarities
  token_post_kernel<<<NTOK / 8, 256, 0, stream>>>(routbuf, outExpr, outCos, penAcc);
  finalize_kernel<<<1, 32, 0, stream>>>(penAcc, lossAcc, outPen, outLoss);
  topk_kernel<<<NTOK / 256, 256, 0, stream>>>(outCos, outPen, ema, outMult, outSel);
}
